// IGCNet_11742440587995
// MI455X (gfx1250) — matrix operand order repaired and checked by reading, not compiled, not run on hardware
//
#include <hip/hip_runtime.h>
#include <hip/hip_fp16.h>

typedef __attribute__((ext_vector_type(16))) _Float16 v16h;
typedef __attribute__((ext_vector_type(8)))  _Float16 v8h;
typedef __attribute__((ext_vector_type(8)))  float    v8f;
typedef __attribute__((ext_vector_type(4)))  float    v4f;

#define NN 50000
#define NE 1600000

// ---------------------------------------------------------------------------
// Pack W1 (12x64) and W2 (64x64) into f16 WMMA B-fragment layout.
// B fragment (16x16x32 f16): lane L holds column n = L%16; half-slot j holds
// K = (j<8 ? j : j+8) + 8*(L/16).  Stored lane-major: frag[f][lane][16 halfs] so each lane
// reads its 32B with contiguous b128 loads.
// W1 fragments: t = N-tile 0..3 (K<12 valid, rest zero).
// W2 fragments: f = kc*4 + t, kc = K-chunk (0: K0..31, 1: K32..63).
// ---------------------------------------------------------------------------
__global__ void pack_weights(const float* __restrict__ W1,
                             const float* __restrict__ W2,
                             _Float16* __restrict__ w1p,
                             _Float16* __restrict__ w2p) {
    int idx = blockIdx.x * blockDim.x + threadIdx.x;
    if (idx < 4 * 32 * 16) {
        int j = idx & 15, lane = (idx >> 4) & 31, t = idx >> 9;
        int k = (j < 8 ? j : j + 8) + 8 * (lane >> 4);
        int col = t * 16 + (lane & 15);
        float v = (k < 12) ? W1[k * 64 + col] : 0.0f;
        w1p[idx] = (_Float16)v;
    }
    if (idx < 8 * 32 * 16) {
        int j = idx & 15, lane = (idx >> 4) & 31, f = idx >> 9;
        int kc = f >> 2, t = f & 3;
        int k = kc * 32 + (j < 8 ? j : j + 8) + 8 * (lane >> 4);
        int col = t * 16 + (lane & 15);
        w2p[idx] = (_Float16)W2[k * 64 + col];
    }
}

// ---------------------------------------------------------------------------
// Edge kernel: each wave processes tiles of 16 edges.
//   A1 (16x32 f16): rows = edges, K = [x[src](8), edge_attr(4), pad 0].
//     16-bit A layout: lanes 0-15 hold K=0..7 (slots 0-7) + K=16..23 (slots
//     8-15); lanes 16-31 hold K=8..15 / K=24..31.  So lanes 0-15 carry x,
//     lanes 16-31 carry edge_attr, the rest is zero pad.
//   H1 = relu(A1@W1+b1) comes out in C-layout; transpose via per-wave LDS
//   tile into A-layout halves for layer 2 (K=64 -> two WMMAs per N-tile).
//   H2 = relu(.@W2+b2) scattered with u32 atomic max (values >= 0).
// ---------------------------------------------------------------------------
__global__ void __launch_bounds__(256)
edge_kernel(const float* __restrict__ xin, const float* __restrict__ eattr,
            const int* __restrict__ srcI, const int* __restrict__ dstI,
            const _Float16* __restrict__ w1p, const _Float16* __restrict__ w2p,
            const float* __restrict__ b1, const float* __restrict__ b2,
            unsigned int* __restrict__ aggr, int numTiles) {
    __shared__ _Float16 h1s[8][16 * 64];
    __shared__ int      dsts[8][16];

    const int lane = threadIdx.x & 31;
    const int w    = threadIdx.x >> 5;
    const int col  = lane & 15;
    const int hi   = lane >> 4;

    // hoist weight fragments + biases into registers for the whole loop
    v16h B1[4], B2[2][4];
#pragma unroll
    for (int t = 0; t < 4; ++t)
        B1[t] = *(const v16h*)(w1p + (t * 32 + lane) * 16);
#pragma unroll
    for (int kc = 0; kc < 2; ++kc)
#pragma unroll
        for (int t = 0; t < 4; ++t)
            B2[kc][t] = *(const v16h*)(w2p + ((kc * 4 + t) * 32 + lane) * 16);

    float bias1[4], bias2[4];
#pragma unroll
    for (int t = 0; t < 4; ++t) {
        bias1[t] = b1[t * 16 + col];
        bias2[t] = b2[t * 16 + col];
    }

    const v8f cz = {0.f, 0.f, 0.f, 0.f, 0.f, 0.f, 0.f, 0.f};
    const int wavesTotal = gridDim.x * 8;

    for (int tile = blockIdx.x * 8 + w; tile < numTiles; tile += wavesTotal) {
        const int e0 = tile * 16;
        int e_lane = e0 + col;
        if (e_lane >= NE) e_lane = NE - 1;   // duplicate edge: max-aggr safe

        v16h a1;
#pragma unroll
        for (int j = 0; j < 16; ++j) a1[j] = (_Float16)0.0f;

        if (hi == 0) {   // lanes 0-15: x[src] -> K=0..7
            int s = srcI[e_lane];
            v4f x0 = *(const v4f*)(xin + (size_t)s * 8);
            v4f x1 = *(const v4f*)(xin + (size_t)s * 8 + 4);
            a1[0] = (_Float16)x0.x; a1[1] = (_Float16)x0.y;
            a1[2] = (_Float16)x0.z; a1[3] = (_Float16)x0.w;
            a1[4] = (_Float16)x1.x; a1[5] = (_Float16)x1.y;
            a1[6] = (_Float16)x1.z; a1[7] = (_Float16)x1.w;
            dsts[w][col] = dstI[e_lane];
        } else {         // lanes 16-31: edge_attr -> K=8..11
            v4f ea = *(const v4f*)(eattr + (size_t)e_lane * 4);
            a1[0] = (_Float16)ea.x; a1[1] = (_Float16)ea.y;
            a1[2] = (_Float16)ea.z; a1[3] = (_Float16)ea.w;
        }

        // ---- layer 1: H1 = relu(A1 @ W1 + b1), 4 N-tiles ----
        v8f acc[4];
#pragma unroll
        for (int t = 0; t < 4; ++t)
            acc[t] = __builtin_amdgcn_wmma_f32_16x16x32_f16(
                false, a1, false, B1[t], (short)0, cz, false, false);

        // bias+relu, C-layout -> LDS row-major f16 tile [16 rows][64 feats]
#pragma unroll
        for (int t = 0; t < 4; ++t)
#pragma unroll
            for (int v = 0; v < 8; ++v) {
                float val = acc[t][v] + bias1[t];
                val = val > 0.f ? val : 0.f;
                int m = v + 8 * hi;
                int n = col + 16 * t;
                h1s[w][m * 64 + n] = (_Float16)val;
            }

        // per-wave LDS ops are in-order; fence compiler + drain DScnt
        asm volatile("s_wait_dscnt 0x0" ::: "memory");

        // ---- reload H1 as A fragments (row = col = my edge) ----
        const _Float16* rowp = &h1s[w][col * 64];
        v16h a2[2];
#pragma unroll
        for (int kc = 0; kc < 2; ++kc) {
            v8h lo  = *(const v8h*)(rowp + kc * 32 + 8 * hi);        // K=base..base+7
            v8h hi8 = *(const v8h*)(rowp + kc * 32 + 16 + 8 * hi);   // K=base+16..+23
            v16h a;
#pragma unroll
            for (int j = 0; j < 8; ++j) { a[j] = lo[j]; a[8 + j] = hi8[j]; }
            a2[kc] = a;
        }

        // ---- layer 2: H2 = relu(H1 @ W2 + b2), K=64 chained ----
#pragma unroll
        for (int t = 0; t < 4; ++t) {
            v8f c = __builtin_amdgcn_wmma_f32_16x16x32_f16(
                false, a2[0], false, B2[0][t], (short)0, cz, false, false);
            c = __builtin_amdgcn_wmma_f32_16x16x32_f16(
                false, a2[1], false, B2[1][t], (short)0, c, false, false);
            acc[t] = c;
        }

        // ---- scatter: segment max via u32 atomic (val >= 0) ----
#pragma unroll
        for (int v = 0; v < 8; ++v) {
            int mm = v + 8 * hi;
            int d  = dsts[w][mm];
#pragma unroll
            for (int t = 0; t < 4; ++t) {
                float val = acc[t][v] + bias2[t];
                val = val > 0.f ? val : 0.f;
                int n = col + 16 * t;
                atomicMax(aggr + (size_t)d * 64 + n, __float_as_uint(val));
            }
        }
    }
}

// ---------------------------------------------------------------------------
// Node kernel: scalar per node (0.24 GFLOP/iter, weight indices uniform ->
// scalar loads).  comb = relu([x,aggr]@W3+b3)@W4+b4; norm clip; concat x[:4].
// ---------------------------------------------------------------------------
__global__ void node_kernel(const float* __restrict__ xin,
                            const float* __restrict__ aggr,
                            const float* __restrict__ W3, const float* __restrict__ b3,
                            const float* __restrict__ W4, const float* __restrict__ b4,
                            float* __restrict__ xout) {
    int i = blockIdx.x * blockDim.x + threadIdx.x;
    if (i >= NN) return;

    float tmp[72];
#pragma unroll
    for (int k = 0; k < 8; ++k) tmp[k] = xin[(size_t)i * 8 + k];
#pragma unroll 8
    for (int k = 0; k < 64; ++k) tmp[8 + k] = aggr[(size_t)i * 64 + k];

    float t32[32];
    for (int c = 0; c < 32; ++c) {
        float s = b3[c];
#pragma unroll 8
        for (int k = 0; k < 72; ++k) s += tmp[k] * W3[k * 32 + c];
        t32[c] = s > 0.f ? s : 0.f;
    }

    float comb[4], nrm = 0.f;
#pragma unroll
    for (int j = 0; j < 4; ++j) {
        float s = b4[j];
#pragma unroll 8
        for (int c = 0; c < 32; ++c) s += t32[c] * W4[c * 4 + j];
        comb[j] = s;
        nrm += s * s;
    }
    nrm = sqrtf(nrm);
    float sc = nrm > 1.f ? 1.f / nrm : 1.f;
#pragma unroll
    for (int j = 0; j < 4; ++j) xout[(size_t)i * 8 + j] = comb[j] * sc;
#pragma unroll
    for (int j = 0; j < 4; ++j) xout[(size_t)i * 8 + 4 + j] = xin[(size_t)i * 8 + j];
}

// ---------------------------------------------------------------------------
extern "C" void kernel_launch(void* const* d_in, const int* in_sizes, int n_in,
                              void* d_out, int out_size, void* d_ws, size_t ws_size,
                              hipStream_t stream) {
    const float* x     = (const float*)d_in[0];
    const float* eattr = (const float*)d_in[1];
    const int*   eidx  = (const int*)d_in[2];
    const float* W1 = (const float*)d_in[3];
    const float* b1 = (const float*)d_in[4];
    const float* W2 = (const float*)d_in[5];
    const float* b2 = (const float*)d_in[6];
    const float* W3 = (const float*)d_in[7];
    const float* b3 = (const float*)d_in[8];
    const float* W4 = (const float*)d_in[9];
    const float* b4 = (const float*)d_in[10];

    const int* srcI = eidx;
    const int* dstI = eidx + NE;

    char* ws = (char*)d_ws;
    unsigned int* aggr = (unsigned int*)ws;                       // NN*64*4 = 12.8 MB
    float* h0 = (float*)(ws + (size_t)NN * 64 * 4);               // NN*8*4
    float* h1 = h0 + (size_t)NN * 8;                              // NN*8*4
    _Float16* w1p = (_Float16*)(h1 + (size_t)NN * 8);             // 2048 halfs
    _Float16* w2p = w1p + 4 * 32 * 16;                            // 4096 halfs

    pack_weights<<<16, 256, 0, stream>>>(W1, W2, w1p, w2p);

    const int numTiles = (NE + 15) / 16;          // 100000 (exact)
    const float* cur = x;
    float* outs[3] = {h0, h1, (float*)d_out};

    for (int it = 0; it < 3; ++it) {
        hipMemsetAsync(aggr, 0, (size_t)NN * 64 * 4, stream);
        edge_kernel<<<2048, 256, 0, stream>>>(cur, eattr, srcI, dstI,
                                              w1p, w2p, b1, b2, aggr, numTiles);
        node_kernel<<<(NN + 255) / 256, 256, 0, stream>>>(cur, (const float*)aggr,
                                                          W3, b3, W4, b4, outs[it]);
        cur = outs[it];
    }
}
